// EnergyBasedSolitonHealer_88304527606077
// MI455X (gfx1250) — matrix operand order repaired and checked by reading, not compiled, not run on hardware
//
#include <hip/hip_runtime.h>

typedef __attribute__((ext_vector_type(16))) _Float16     v16h;
typedef __attribute__((ext_vector_type(8)))  float        v8f;
typedef __attribute__((ext_vector_type(8)))  unsigned int v8u;
typedef __attribute__((ext_vector_type(2)))  __fp16       f16x2;  // matches cvt_pkrtz return type

#define HEALING_RATE  0.1f
#define ENERGY_MARGIN 1.0f
#define STAB2         1e-6f   // (1e-3)^2, compare squared norm

__device__ __forceinline__ unsigned int pk16(float a, float b) {
  f16x2 p = __builtin_amdgcn_cvt_pkrtz(a, b);   // v_cvt_pk_rtz_f16_f32
  return __builtin_bit_cast(unsigned int, p);
}

// One wave heals 16 samples; state lives in registers across all steps.
// O-layout (WMMA C/D layout): lane n=lane&15 -> sample, h=lane>>4,
// reg (t,r) -> dim = 16*t + 8*h + r.
__global__ __launch_bounds__(128) void soliton_heal_wmma(
    const float* __restrict__ state, const float* __restrict__ W,
    const float* __restrict__ bias,  const float* __restrict__ tmpl,
    const int*   __restrict__ iters, float* __restrict__ out)
{
  const int lane = threadIdx.x & 31;
  const int wave = threadIdx.x >> 5;
  const int h    = lane >> 4;
  const int n    = lane & 15;
  const long long row = (long long)blockIdx.x * 64ll + wave * 16 + n;
  const float* srow = state + row * 64;
  float*       orow = out   + row * 64;

  // ---- constants in O-layout ----
  float tgt[4][8], bia[4][8];
#pragma unroll
  for (int t = 0; t < 4; ++t)
#pragma unroll
    for (int r = 0; r < 8; ++r) {
      const int d = 16 * t + 8 * h + r;
      tgt[t][r] = tmpl[d];
      bia[t][r] = bias[d];
    }

  // ---- W_sym = W + W^T as f16 WMMA A-fragments (loaded once, reused all steps) ----
  // A layout (16x32 f16): M = mt*16 + n; pair i: K = 32c + 16h + 2*i + {0,1}
  v16h wA[4][2];
#pragma unroll
  for (int mt = 0; mt < 4; ++mt) {
    const int m = mt * 16 + n;
#pragma unroll
    for (int c = 0; c < 2; ++c) {
      v8u au;
#pragma unroll
      for (int i = 0; i < 8; ++i) {
        const int k0 = 32 * c + 16 * h + 2 * i;
        const float2 wr = *(const float2*)(W + m * 64 + k0);
        const float a0 = wr.x + W[k0 * 64 + m];
        const float a1 = wr.y + W[(k0 + 1) * 64 + m];
        au[i] = pk16(a0, a1);
      }
      wA[mt][c] = __builtin_bit_cast(v16h, au);
    }
  }

  // ---- load state once (8x global_load_b128 per lane) ----
  float s[4][8];
#pragma unroll
  for (int t = 0; t < 4; ++t) {
    const float4 lo = *(const float4*)(srow + 16 * t + 8 * h);
    const float4 hi = *(const float4*)(srow + 16 * t + 8 * h + 4);
    s[t][0] = lo.x; s[t][1] = lo.y; s[t][2] = lo.z; s[t][3] = lo.w;
    s[t][4] = hi.x; s[t][5] = hi.y; s[t][6] = hi.z; s[t][7] = hi.w;
  }

  const int nsteps = iters[0] * 10;   // device-side read (graph-capture safe)
  bool done = false;

  for (int step = 0; step < nsteps; ++step) {
    // d = s - target, packed to f16 pairs
    float d[4][8];
#pragma unroll
    for (int t = 0; t < 4; ++t)
#pragma unroll
      for (int r = 0; r < 8; ++r) d[t][r] = s[t][r] - tgt[t][r];

    unsigned int dpk[4][4], dpo[4][4];
#pragma unroll
    for (int t = 0; t < 4; ++t)
#pragma unroll
      for (int i = 0; i < 4; ++i) {
        dpk[t][i] = pk16(d[t][2 * i], d[t][2 * i + 1]);
        dpo[t][i] = (unsigned int)__shfl_xor((int)dpk[t][i], 16, 32);
      }

    // B fragments (32x16 f16): N = n, element j -> K = kb + 16h + j.
    // Elements whose dims live in the partner lane come from the xor-16 shuffle.
    v8u b0u, b1u;
#pragma unroll
    for (int i = 0; i < 4; ++i) {
      b0u[i]     = h ? dpo[1][i] : dpk[0][i];
      b0u[4 + i] = h ? dpk[1][i] : dpo[0][i];
      b1u[i]     = h ? dpo[3][i] : dpk[2][i];
      b1u[4 + i] = h ? dpk[3][i] : dpo[2][i];
    }
    const v16h B0 = __builtin_bit_cast(v16h, b0u);
    const v16h B1 = __builtin_bit_cast(v16h, b1u);

    // g0^T = W_sym x d^T : 8 WMMA per step per wave (K=64 in two chunks)
    float g[4][8];
    float edg = 0.f, esb = 0.f, gn2 = 0.f;
#pragma unroll
    for (int mt = 0; mt < 4; ++mt) {
      v8f acc = {};
      acc = __builtin_amdgcn_wmma_f32_16x16x32_f16(false, wA[mt][0], false, B0,
                                                   (short)0, acc, false, false);
      acc = __builtin_amdgcn_wmma_f32_16x16x32_f16(false, wA[mt][1], false, B1,
                                                   (short)0, acc, false, false);
#pragma unroll
      for (int r = 0; r < 8; ++r) {
        const float g0 = acc[r];
        const float gv = g0 + bia[mt][r];
        g[mt][r] = gv;
        edg = fmaf(d[mt][r], g0, edg);          // Σ d.(d@W_sym)  (x0.5 applied once below)
        esb = fmaf(s[mt][r], bia[mt][r], esb);  // Σ s.b
        gn2 = fmaf(gv, gv, gn2);
      }
    }
    // E = 0.5 * Σ d.(d@W_sym) + Σ s.b ;  lane pair holds disjoint dim halves
    float eP = fmaf(0.5f, edg, esb);
    eP  += __shfl_xor(eP,  16, 32);
    gn2 += __shfl_xor(gn2, 16, 32);

    // done-freeze via rate: once done, s is already clipped, so clip(s - 0*g) == s.
    const float rate = done ? 0.f
                            : ((eP < ENERGY_MARGIN) ? (0.1f * HEALING_RATE) : HEALING_RATE);
#pragma unroll
    for (int t = 0; t < 4; ++t)
#pragma unroll
      for (int r = 0; r < 8; ++r)
        s[t][r] = fminf(fmaxf(fmaf(-rate, g[t][r], s[t][r]), -10.f), 10.f);
    done = done || (gn2 < STAB2);   // break-after-update semantics
  }

  // ---- store once (8x global_store_b128 per lane) ----
#pragma unroll
  for (int t = 0; t < 4; ++t) {
    const float4 lo = make_float4(s[t][0], s[t][1], s[t][2], s[t][3]);
    const float4 hi = make_float4(s[t][4], s[t][5], s[t][6], s[t][7]);
    *(float4*)(orow + 16 * t + 8 * h)     = lo;
    *(float4*)(orow + 16 * t + 8 * h + 4) = hi;
  }
}

extern "C" void kernel_launch(void* const* d_in, const int* in_sizes, int n_in,
                              void* d_out, int out_size, void* d_ws, size_t ws_size,
                              hipStream_t stream) {
  const float* state = (const float*)d_in[0];   // [BATCH, 64] f32
  const float* W     = (const float*)d_in[1];   // [64, 64]    f32
  const float* bias  = (const float*)d_in[2];   // [64]        f32
  const float* tmpl  = (const float*)d_in[3];   // [64]        f32
  const int*   itc   = (const int*)d_in[4];     // scalar
  float* out = (float*)d_out;

  const int batch = in_sizes[0] / 64;           // 524288
  const int grid  = batch / 64;                 // 64 rows per 128-thread block (4 waves x 16)
  soliton_heal_wmma<<<grid, 128, 0, stream>>>(state, W, bias, tmpl, itc, out);
}
